// SelfAttention_81355270521413
// MI455X (gfx1250) — matrix operand order repaired and checked by reading, not compiled, not run on hardware
//
#include <hip/hip_runtime.h>
#include <hip/hip_bf16.h>

// ---------------------------------------------------------------------------
// Shapes (compile-time constants for this problem)
// ---------------------------------------------------------------------------
#define BATCH   2
#define SEQ     2048
#define DMODEL  1024
#define NHEADS  16
#define HDIM    64
#define MROWS   (BATCH * SEQ)      // 4096
#define NEG_INF (-3.0e38f)

typedef __attribute__((ext_vector_type(16))) __bf16 v16bf;
typedef __attribute__((ext_vector_type(8)))  __bf16 v8bf;
typedef __attribute__((ext_vector_type(8)))  float  v8f;
typedef __attribute__((ext_vector_type(4)))  unsigned int u32x4;
typedef __attribute__((ext_vector_type(8)))  int          i32x8;
typedef __attribute__((ext_vector_type(4)))  int          i32x4;

// ---------------------------------------------------------------------------
// CDNA5 async global->LDS copy (ASYNCcnt path).  LDS byte offset is the low
// 32 bits of the generic pointer.
// ---------------------------------------------------------------------------
static __device__ __forceinline__ void async_g2l_b128(const void* gptr,
                                                      void* lptr) {
  asm volatile("global_load_async_to_lds_b128 %0, %1, off"
               :
               : "v"((unsigned int)(size_t)lptr),
                 "v"((unsigned long long)(size_t)gptr)
               : "memory");
}
static __device__ __forceinline__ void wait_async0() {
  asm volatile("s_wait_asynccnt 0" ::: "memory");
}

// ---------------------------------------------------------------------------
// CDNA5 Tensor Data Mover: 2D tile load (bf16 elements), TENSORcnt path.
// D# per CDNA5 ISA: group0 = {count/type, lds_addr,
// global_addr}, group1 = {data_size, tensor dims, tile dims, strides}.
// ---------------------------------------------------------------------------
static __device__ __forceinline__ void tdm_load_2d_bf16(
    unsigned ldsoff, const void* g, unsigned tensor_d0, unsigned tensor_d1,
    unsigned tile_d0, unsigned tile_d1, unsigned long long d0_stride) {
  unsigned long long ga = (unsigned long long)(size_t)g;
  u32x4 g0;
  g0.x = 1u;                                            // count=1, is_restore=0
  g0.y = ldsoff;                                        // lds_addr
  g0.z = (unsigned)(ga & 0xffffffffu);                  // global_addr[31:0]
  g0.w = (unsigned)((ga >> 32) & 0x01ffffffu)           // global_addr[56:32]
         | 0x80000000u;                                 // type=2 ("image")
  i32x8 g1;
  g1[0] = 0x10000;                                      // data_size=1 (2 bytes)
  g1[1] = (int)((tensor_d0 & 0xffffu) << 16);           // tensor_dim0[15:0]
  g1[2] = (int)(((tensor_d0 >> 16) & 0xffffu) |
                ((tensor_d1 & 0xffffu) << 16));         // d0[31:16] | d1[15:0]
  g1[3] = (int)(((tensor_d1 >> 16) & 0xffffu) |
                ((tile_d0 & 0xffffu) << 16));           // d1[31:16] | tile0
  g1[4] = (int)(tile_d1 & 0xffffu);                     // tile_dim1, tile_dim2=0
  g1[5] = (int)(d0_stride & 0xffffffffu);               // tensor_dim0_stride lo
  g1[6] = (int)((d0_stride >> 32) & 0xffffu);           // stride hi, d1stride=0
  g1[7] = 0;
  i32x4 z = {0, 0, 0, 0};
#if defined(__clang_major__) && (__clang_major__ >= 23)
  i32x8 z8 = {0, 0, 0, 0, 0, 0, 0, 0};
  __builtin_amdgcn_tensor_load_to_lds(g0, g1, z, z, z8, 0);
#else
  __builtin_amdgcn_tensor_load_to_lds(g0, g1, z, z, 0);
#endif
}

// ---------------------------------------------------------------------------
// WMMA fragment helpers (CDNA5 16x16x32 bf16 VGPR layouts, wave32)
//   A (16x32): lane<16 -> row=lane, K = {lhi*8..+7} U {16+lhi*8..+7}
//   B (32x16): col = lane&15, K = {lhi*8..+7} U {16+lhi*8..+7}
// ---------------------------------------------------------------------------
static __device__ __forceinline__ v16bf combine8(v8bf lo, v8bf hi) {
  v16bf r;
#pragma unroll
  for (int e = 0; e < 8; ++e) { r[e] = lo[e]; r[8 + e] = hi[e]; }
  return r;
}

static __device__ __forceinline__ v16bf frag_a(const __bf16* row, int lhi) {
  v8bf lo = *(const v8bf*)(row + lhi * 8);
  v8bf hi = *(const v8bf*)(row + 16 + lhi * 8);
  return combine8(lo, hi);
}

static __device__ __forceinline__ v16bf frag_b(const __bf16* row, int lhi) {
  v8bf lo = *(const v8bf*)(row + lhi * 8);
  v8bf hi = *(const v8bf*)(row + 16 + lhi * 8);
  return combine8(lo, hi);
}

static __device__ __forceinline__ v8f wmma_bf16(v16bf a, v16bf b, v8f c) {
  return __builtin_amdgcn_wmma_f32_16x16x32_bf16(false, a, false, b, (short)0,
                                                 c, false, false);
}

// ---------------------------------------------------------------------------
// fp32 -> bf16 conversion
// ---------------------------------------------------------------------------
__global__ void cvt_f32_to_bf16(const float* __restrict__ in,
                                __bf16* __restrict__ out, int n) {
  int i = blockIdx.x * blockDim.x + threadIdx.x;
  int stride = gridDim.x * blockDim.x;
  for (; i < n; i += stride) out[i] = (__bf16)in[i];
}

// ---------------------------------------------------------------------------
// GEMM: C[m,n] = (sum_k A[m,k] * W[n,k] + bias[n]) * scale
//   A: [MROWS, 1024] bf16 row-major, W: [1024, 1024] bf16 row-major; X @ W^T
//   keeps both operands K-contiguous -> native WMMA fragment layouts.
//   Double-buffered async global->LDS pipeline: the DMA for tile t+1 runs
//   while WMMA consumes tile t; ASYNCcnt is drained only at iteration end.
//   mode 0: store bf16 into [B, H, S, hd]; mode 1: store fp32 [MROWS, 1024].
// Block: 256 threads (8 waves), tile 128x128; wave tile 64(M) x 32(N).
// ---------------------------------------------------------------------------
__global__ __launch_bounds__(256)
void gemm_xwT_bf16(const __bf16* __restrict__ A, const __bf16* __restrict__ W,
                   const float* __restrict__ bias, float scale,
                   void* __restrict__ outp, int mode) {
  const int K = DMODEL, N = DMODEL;
  // 32 cols padded to 40 (80B rows keep every 16B chunk aligned)
  __shared__ __bf16 As[2][128 * 40];
  __shared__ __bf16 Bs[2][128 * 40];

  const int m0 = blockIdx.y * 128;
  const int n0 = blockIdx.x * 128;
  const int tid  = threadIdx.x;
  const int wave = tid >> 5;
  const int lane = tid & 31;
  const int llo  = lane & 15;
  const int lhi  = lane >> 4;
  const int wm   = wave & 1;        // 2 wave rows  -> 64 M each
  const int wn   = wave >> 1;       // 4 wave cols  -> 32 N each

  auto issue_tile = [&](int k0, int buf) {
#pragma unroll
    for (int c = tid; c < 512; c += 256) {   // 512 chunks of 16B per tile
      int r = c >> 2, s = c & 3;
      async_g2l_b128(A + (size_t)(m0 + r) * K + k0 + s * 8,
                     &As[buf][r * 40 + s * 8]);
      async_g2l_b128(W + (size_t)(n0 + r) * K + k0 + s * 8,
                     &Bs[buf][r * 40 + s * 8]);
    }
  };

  v8f acc[4][2] = {};

  issue_tile(0, 0);
  wait_async0();
  __syncthreads();

  for (int k0 = 0; k0 < K; k0 += 32) {
    const int cur = (k0 >> 5) & 1;

    if (k0 + 32 < K) {
      issue_tile(k0 + 32, cur ^ 1);            // overlap DMA with WMMA below
      // pull the tile after next toward the WGP (nearest-scope prefetch)
      if (k0 + 64 < K)
        __builtin_prefetch(A + (size_t)(m0 + (tid >> 1)) * K + k0 + 64 +
                               (tid & 1) * 16, 0, 3);
    }

    v16bf bfrag[2];
#pragma unroll
    for (int j = 0; j < 2; ++j)
      bfrag[j] = frag_b(&Bs[cur][(wn * 32 + j * 16 + llo) * 40], lhi);
#pragma unroll
    for (int i = 0; i < 4; ++i) {
      v16bf afrag = frag_a(&As[cur][(wm * 64 + i * 16 + llo) * 40], lhi);
#pragma unroll
      for (int j = 0; j < 2; ++j)
        acc[i][j] = wmma_bf16(afrag, bfrag[j], acc[i][j]);
    }

    wait_async0();      // next tile's copies have landed (mostly overlapped)
    __syncthreads();    // includes DScnt drain: reads of 'cur' are done
  }

  // Epilogue: C/D layout -> row M = lhi*8 + r, col N = llo
#pragma unroll
  for (int i = 0; i < 4; ++i) {
#pragma unroll
    for (int j = 0; j < 2; ++j) {
#pragma unroll
      for (int r = 0; r < 8; ++r) {
        int gm = m0 + wm * 64 + i * 16 + lhi * 8 + r;
        int gn = n0 + wn * 32 + j * 16 + llo;
        float v = (acc[i][j][r] + bias[gn]) * scale;
        if (mode == 0) {
          int b = gm >> 11, s = gm & (SEQ - 1);
          int h = gn >> 6,  d = gn & (HDIM - 1);
          ((__bf16*)outp)[(((size_t)(b * NHEADS + h)) * SEQ + s) * HDIM + d] =
              (__bf16)v;
        } else {
          ((float*)outp)[(size_t)gm * N + gn] = v;
        }
      }
    }
  }
}

// ---------------------------------------------------------------------------
// Flash attention (causal), bf16 WMMA, fp32 online softmax.
// Grid: (S/128, H, B); block 256 threads = 8 waves; each wave owns 16 queries.
// K tile (64x64) fetched by the Tensor Data Mover (TENSORcnt); V transposed
// into LDS for the PV B-operand.
// ---------------------------------------------------------------------------
__global__ __launch_bounds__(256)
void flash_attn_bf16(const __bf16* __restrict__ Q, const __bf16* __restrict__ K,
                     const __bf16* __restrict__ V, __bf16* __restrict__ Aout) {
  __shared__ __bf16 Kt[64 * HDIM];      // [key][d]  (natural: B-op of Q.K^T)
  __shared__ __bf16 Vt[HDIM * 64];      // [d][key]  (transposed: B-op of P.V)
  __shared__ __bf16 Pt[8][16 * 64];     // per-wave P staging (C->A relayout)

  const int qb = blockIdx.x, h = blockIdx.y, b = blockIdx.z;
  const int tid  = threadIdx.x;
  const int wave = tid >> 5;
  const int lane = tid & 31;
  const int llo  = lane & 15;
  const int lhi  = lane >> 4;
  const int qbase = qb * 128;

  const size_t head = (size_t)(b * NHEADS + h) * SEQ * HDIM;
  const __bf16* Qh = Q + head;
  const __bf16* Kh = K + head;
  const __bf16* Vh = V + head;

  // Q fragments live in registers for the whole kernel (A-operand layout).
  const __bf16* qrow = Qh + (size_t)(qbase + wave * 16 + llo) * HDIM;
  v16bf aq0 = frag_a(qrow, lhi);        // d = 0..31
  v16bf aq1 = frag_a(qrow + 32, lhi);   // d = 32..63

  float m_i[8], l_i[8];
  v8f   o_acc[4] = {};
#pragma unroll
  for (int r = 0; r < 8; ++r) { m_i[r] = NEG_INF; l_i[r] = 0.0f; }

  const int nkv = qb * 2 + 2;           // causal: keys up to qbase+127
  for (int blk = 0; blk < nkv; ++blk) {
    const int kv0 = blk * 64;
    __syncthreads();

    // K tile via TDM: tensor [SEQ x 64] bf16, tile 64x64 at row kv0.
    if (wave == 0) {
      tdm_load_2d_bf16((unsigned)(size_t)&Kt[0], Kh + (size_t)kv0 * HDIM,
                       HDIM, SEQ, HDIM, 64, HDIM);
    }

    {   // V tile, transposed into LDS: thread -> (row, 16-element segment)
      int r = tid >> 2, s = tid & 3;
      const __bf16* vs = Vh + (size_t)(kv0 + r) * HDIM + s * 16;
      v8bf v0 = *(const v8bf*)vs, v1 = *(const v8bf*)(vs + 8);
#pragma unroll
      for (int e = 0; e < 8; ++e) {
        Vt[(s * 16 + e) * 64 + r]     = v0[e];
        Vt[(s * 16 + 8 + e) * 64 + r] = v1[e];
      }
      if (blk + 1 < nkv) {  // nearest-scope prefetch of the next KV tiles
        __builtin_prefetch(Kh + (size_t)(kv0 + 64 + r) * HDIM + s * 16, 0, 3);
        __builtin_prefetch(Vh + (size_t)(kv0 + 64 + r) * HDIM + s * 16, 0, 3);
      }
    }

    if (wave == 0) __builtin_amdgcn_s_wait_tensorcnt(0);
    __syncthreads();

    // scores: 16 queries x 64 keys per wave
    v8f cs[4];
#pragma unroll
    for (int j = 0; j < 4; ++j) {
      const __bf16* krow = &Kt[(j * 16 + llo) * HDIM];
      v8f z = {};
      z = wmma_bf16(aq0, frag_b(krow, lhi), z);
      z = wmma_bf16(aq1, frag_b(krow + 32, lhi), z);
      cs[j] = z;
    }

    // causal mask + online softmax (row = lhi*8 + r, key col = llo + 16j)
#pragma unroll
    for (int r = 0; r < 8; ++r) {
      int q = qbase + wave * 16 + lhi * 8 + r;
      float bmax = NEG_INF;
#pragma unroll
      for (int j = 0; j < 4; ++j) {
        int key = kv0 + j * 16 + llo;
        if (key > q) cs[j][r] = NEG_INF;
        bmax = fmaxf(bmax, cs[j][r]);
      }
#pragma unroll
      for (int off = 1; off < 16; off <<= 1)
        bmax = fmaxf(bmax, __shfl_xor(bmax, off, 32));
      float nm    = fmaxf(m_i[r], bmax);
      float alpha = __expf(m_i[r] - nm);
      float psum  = 0.0f;
#pragma unroll
      for (int j = 0; j < 4; ++j) {
        float p = __expf(cs[j][r] - nm);
        psum += p;
        Pt[wave][(lhi * 8 + r) * 64 + j * 16 + llo] = (__bf16)p;
      }
#pragma unroll
      for (int off = 1; off < 16; off <<= 1)
        psum += __shfl_xor(psum, off, 32);
      l_i[r] = l_i[r] * alpha + psum;
      m_i[r] = nm;
#pragma unroll
      for (int n = 0; n < 4; ++n) o_acc[n][r] *= alpha;
    }

    // P stores -> A-fragment loads are same-wave LDS; DS is in-order, but
    // make the dependency explicit with the CDNA5 split counter wait.
    asm volatile("s_wait_dscnt 0" ::: "memory");

    // O += P @ V   (A = P over keys; B = Vt[d][key])
#pragma unroll
    for (int t = 0; t < 2; ++t) {
      v16bf ap = frag_a(&Pt[wave][llo * 64 + t * 32], lhi);
#pragma unroll
      for (int n = 0; n < 4; ++n) {
        const __bf16* vrow = &Vt[(n * 16 + llo) * 64 + t * 32];
        o_acc[n] = wmma_bf16(ap, frag_b(vrow, lhi), o_acc[n]);
      }
    }
  }

  // normalize + write attn output as bf16 [B, S, D]
#pragma unroll
  for (int r = 0; r < 8; ++r) {
    int q = qbase + wave * 16 + lhi * 8 + r;
    float inv = 1.0f / l_i[r];
#pragma unroll
    for (int n = 0; n < 4; ++n) {
      float v = o_acc[n][r] * inv;
      Aout[((size_t)(b * SEQ + q)) * DMODEL + h * HDIM + n * 16 + llo] =
          (__bf16)v;
    }
  }
}

// ---------------------------------------------------------------------------
// Host-side launch
// inputs: 0 hidden_states f32 [B,S,D], 1 mask (unused), 2 Wq, 3 bq, 4 Wk,
//         5 bk, 6 Wv, 7 bv, 8 Wo, 9 bo
// ---------------------------------------------------------------------------
extern "C" void kernel_launch(void* const* d_in, const int* in_sizes, int n_in,
                              void* d_out, int out_size, void* d_ws,
                              size_t ws_size, hipStream_t stream) {
  (void)in_sizes; (void)n_in; (void)out_size; (void)ws_size;
  const float* hs = (const float*)d_in[0];
  const float* Wq = (const float*)d_in[2]; const float* bq = (const float*)d_in[3];
  const float* Wk = (const float*)d_in[4]; const float* bk = (const float*)d_in[5];
  const float* Wv = (const float*)d_in[6]; const float* bv = (const float*)d_in[7];
  const float* Wo = (const float*)d_in[8]; const float* bo = (const float*)d_in[9];
  float* out = (float*)d_out;

  const size_t XD = (size_t)MROWS * DMODEL;     // 4096*1024
  const size_t WD = (size_t)DMODEL * DMODEL;    // 1024*1024
  __bf16* Xb  = (__bf16*)d_ws;
  __bf16* Wqb = Xb  + XD;
  __bf16* Wkb = Wqb + WD;
  __bf16* Wvb = Wkb + WD;
  __bf16* Wob = Wvb + WD;
  __bf16* Qb  = Wob + WD;   // [B,H,S,hd]
  __bf16* Kb  = Qb  + XD;
  __bf16* Vb  = Kb  + XD;
  __bf16* Ab  = Vb  + XD;   // [B,S,D]

  cvt_f32_to_bf16<<<1024, 256, 0, stream>>>(hs, Xb,  (int)XD);
  cvt_f32_to_bf16<<<1024, 256, 0, stream>>>(Wq, Wqb, (int)WD);
  cvt_f32_to_bf16<<<1024, 256, 0, stream>>>(Wk, Wkb, (int)WD);
  cvt_f32_to_bf16<<<1024, 256, 0, stream>>>(Wv, Wvb, (int)WD);
  cvt_f32_to_bf16<<<1024, 256, 0, stream>>>(Wo, Wob, (int)WD);

  dim3 gg(DMODEL / 128, MROWS / 128);           // (8, 32)
  const float scaling = 0.125f;                 // 64^-0.5
  gemm_xwT_bf16<<<gg, 256, 0, stream>>>(Xb, Wqb, bq, scaling, Qb, 0);
  gemm_xwT_bf16<<<gg, 256, 0, stream>>>(Xb, Wkb, bk, 1.0f,    Kb, 0);
  gemm_xwT_bf16<<<gg, 256, 0, stream>>>(Xb, Wvb, bv, 1.0f,    Vb, 0);

  flash_attn_bf16<<<dim3(SEQ / 128, NHEADS, BATCH), 256, 0, stream>>>(Qb, Kb,
                                                                      Vb, Ab);

  gemm_xwT_bf16<<<gg, 256, 0, stream>>>(Ab, Wob, bo, 1.0f, out, 1);
}
